// TorchLMHeadJSD_4183298146644
// MI455X (gfx1250) — matrix operand order repaired and checked by reading, not compiled, not run on hardware
//
#include <hip/hip_runtime.h>
#include <hip/hip_bf16.h>
#include <float.h>

// ---------------------------------------------------------------------------
// Fused LM-head JSD for MI455X (gfx1250, wave32, WMMA).
//   logits_s = Xs @ Ws^T ; logits_t = Xt @ Wt^T   (bf16 WMMA, f32 accum)
//   logits stored bf16 in workspace; per-row online softmax stats; JSD(0.5).
// HBM traffic budget: W read once (1.05 GB), X L2-resident (67 MB),
// logits 262 MB write + 524 MB read  =>  ~1.9 GB  => ~82 us at 23.3 TB/s.
// Grid order puts the 16 M-tiles sharing one W tile adjacent in launch order
// so W is single-read; full X (67 MB) lives in the 192 MB L2 across its
// 250x reuse.
// ---------------------------------------------------------------------------

typedef __attribute__((ext_vector_type(16))) __bf16 bf16x16;
typedef __attribute__((ext_vector_type(4)))  __bf16 bf16x4;
typedef __attribute__((ext_vector_type(8)))  float  f32x8;

#define TILE_M 128
#define TILE_N 128
#define TILE_K 32
#define LDS_STRIDE 48   // 32 + 16 pad; 96 B/row keeps 16B alignment, skews banks

__global__ __launch_bounds__(256)
void lmhead_gemm_bf16wmma(const float* __restrict__ Xs,
                          const float* __restrict__ Ws,
                          const float* __restrict__ Xt,
                          const float* __restrict__ Wt,
                          __bf16* __restrict__ outS,
                          __bf16* __restrict__ outT,
                          int N, int V, int H)
{
    // double-buffered stages: 2 * (12 KB + 12 KB) = 48 KB
    __shared__ __bf16 As[2][TILE_M * LDS_STRIDE];
    __shared__ __bf16 Bs[2][TILE_N * LDS_STRIDE];

    const float* X;
    const float* W;
    __bf16* out;
    if (blockIdx.z == 0) { X = Xs; W = Ws; out = outS; }
    else                 { X = Xt; W = Wt; out = outT; }

    const int tid  = threadIdx.x;
    const int lane = tid & 31;
    const int wave = tid >> 5;      // 0..7 (8 waves of 32)
    const int wm   = wave & 3;      // wave row   (4)
    const int wn   = wave >> 2;     // wave col   (2)
    const int l15  = lane & 15;
    const bool hi  = (lane >= 16);

    const int tileM = blockIdx.x * TILE_M;   // x fastest: M-tiles share W tile
    const int tileN = blockIdx.y * TILE_N;

    f32x8 acc[2][4];
    #pragma unroll
    for (int mi = 0; mi < 2; ++mi)
        #pragma unroll
        for (int ni = 0; ni < 4; ++ni)
            acc[mi][ni] = (f32x8){0.f,0.f,0.f,0.f,0.f,0.f,0.f,0.f};

    // global->LDS staging: each of 256 threads moves 16 floats of A and B
    const int ldRow = tid >> 1;          // 0..127
    const int ldCol = (tid & 1) * 16;    // 0 or 16
    const float* baseA = X + (size_t)(tileM + ldRow) * H + ldCol;
    const float* baseB = W + (size_t)(tileN + ldRow) * H + ldCol;

    const int steps = H / TILE_K;
    float4 ra[4], rb[4];

    // prologue: fetch k-step 0 and stage into buffer 0
    #pragma unroll
    for (int j = 0; j < 4; ++j) {
        ra[j] = ((const float4*)baseA)[j];
        rb[j] = ((const float4*)baseB)[j];
    }
    #pragma unroll
    for (int j = 0; j < 4; ++j) {
        __bf16* ap = &As[0][ldRow * LDS_STRIDE + ldCol + j * 4];
        __bf16* bp = &Bs[0][ldRow * LDS_STRIDE + ldCol + j * 4];
        ap[0] = (__bf16)ra[j].x; ap[1] = (__bf16)ra[j].y;
        ap[2] = (__bf16)ra[j].z; ap[3] = (__bf16)ra[j].w;
        bp[0] = (__bf16)rb[j].x; bp[1] = (__bf16)rb[j].y;
        bp[2] = (__bf16)rb[j].z; bp[3] = (__bf16)rb[j].w;
    }

    for (int kt = 0; kt < steps; ++kt) {
        __syncthreads();
        const int cur = kt & 1;

        // issue global loads for the next k-step (overlap with WMMA below)
        if (kt + 1 < steps) {
            const float* sA = baseA + (kt + 1) * TILE_K;
            const float* sB = baseB + (kt + 1) * TILE_K;
            #pragma unroll
            for (int j = 0; j < 4; ++j) {
                ra[j] = ((const float4*)sA)[j];
                rb[j] = ((const float4*)sB)[j];
            }
        }

        // A fragments: 16x32 bf16. lane<16: row=l15, K 0..7 & 16..23
        //                          lane>=16: row=l15, K 8..15 & 24..31
        bf16x16 afr[2];
        #pragma unroll
        for (int mi = 0; mi < 2; ++mi) {
            const int row = wm * 32 + mi * 16 + l15;
            const int kb  = hi ? 8 : 0;
            const __bf16* p = &As[cur][row * LDS_STRIDE];
            #pragma unroll
            for (int j = 0; j < 8; ++j) {
                afr[mi][j]     = p[kb + j];
                afr[mi][8 + j] = p[16 + kb + j];
            }
        }
        // B fragments: 32x16 bf16 (K x N), B[k][n] = W[tileN+n][k0+k].
        // lanes 0-15: K 0..7 & 16..23 ; lanes 16-31: K 8..15 & 24..31 ; N = l15.
        bf16x16 bfr[4];
        #pragma unroll
        for (int ni = 0; ni < 4; ++ni) {
            const int col = wn * 64 + ni * 16 + l15;
            const int kb  = hi ? 8 : 0;
            const __bf16* p = &Bs[cur][col * LDS_STRIDE + kb];
            #pragma unroll
            for (int e = 0; e < 8; ++e) { bfr[ni][e] = p[e]; bfr[ni][8 + e] = p[16 + e]; }
        }

        #pragma unroll
        for (int mi = 0; mi < 2; ++mi)
            #pragma unroll
            for (int ni = 0; ni < 4; ++ni)
                acc[mi][ni] = __builtin_amdgcn_wmma_f32_16x16x32_bf16(
                    false, afr[mi], false, bfr[ni],
                    (short)0, acc[mi][ni], false, false);

        // stage next k-step into the other buffer
        if (kt + 1 < steps) {
            const int nxt = cur ^ 1;
            #pragma unroll
            for (int j = 0; j < 4; ++j) {
                __bf16* ap = &As[nxt][ldRow * LDS_STRIDE + ldCol + j * 4];
                __bf16* bp = &Bs[nxt][ldRow * LDS_STRIDE + ldCol + j * 4];
                ap[0] = (__bf16)ra[j].x; ap[1] = (__bf16)ra[j].y;
                ap[2] = (__bf16)ra[j].z; ap[3] = (__bf16)ra[j].w;
                bp[0] = (__bf16)rb[j].x; bp[1] = (__bf16)rb[j].y;
                bp[2] = (__bf16)rb[j].z; bp[3] = (__bf16)rb[j].w;
            }
        }
    }

    // D layout: VGPR r -> row r (lanes 0-15) / row r+8 (lanes 16-31), col = l15
    #pragma unroll
    for (int mi = 0; mi < 2; ++mi) {
        #pragma unroll
        for (int ni = 0; ni < 4; ++ni) {
            const int col   = tileN + wn * 64 + ni * 16 + l15;
            const int rbase = tileM + wm * 32 + mi * 16 + (hi ? 8 : 0);
            #pragma unroll
            for (int r = 0; r < 8; ++r)
                out[(size_t)(rbase + r) * V + col] = (__bf16)acc[mi][ni][r];
        }
    }
}

// ---------------------------------------------------------------------------
// Per-row JSD on bf16 logits: pass 1 = online (max, sumexp) for both rows,
// pass 2 = beta*KL(p||m) + (1-beta)*KL(q||m), beta = 0.5.
// b64 vector loads (4 x bf16 per lane), scalar remainder.
// ---------------------------------------------------------------------------
__global__ __launch_bounds__(256)
void jsd_row_kernel(const __bf16* __restrict__ logS,
                    const __bf16* __restrict__ logT,
                    const int* __restrict__ label,
                    float* __restrict__ rowLoss,
                    int V)
{
    const int n   = blockIdx.x;
    const int tid = threadIdx.x;
    const __bf16* ls = logS + (size_t)n * V;
    const __bf16* lt = logT + (size_t)n * V;

    __shared__ float sMs[256], sSs[256], sMt[256], sSt[256];

    const int fullIters = V >> 10;          // 256 threads * 4 elems = 1024/iter

    float ms = -FLT_MAX, ss = 0.f;
    float mt = -FLT_MAX, st = 0.f;
    for (int it = 0; it < fullIters; ++it) {
        const int j = (it << 10) + (tid << 2);
        bf16x4 vs = *(const bf16x4*)(ls + j);
        bf16x4 vt = *(const bf16x4*)(lt + j);
        #pragma unroll
        for (int e = 0; e < 4; ++e) {
            float xs = (float)vs[e];
            float nm = fmaxf(ms, xs);
            ss = ss * __expf(ms - nm) + __expf(xs - nm);
            ms = nm;
            float xt = (float)vt[e];
            float nt = fmaxf(mt, xt);
            st = st * __expf(mt - nt) + __expf(xt - nt);
            mt = nt;
        }
    }
    for (int j = (fullIters << 10) + tid; j < V; j += 256) {
        float xs = (float)ls[j];
        float nm = fmaxf(ms, xs);
        ss = ss * __expf(ms - nm) + __expf(xs - nm);
        ms = nm;
        float xt = (float)lt[j];
        float nt = fmaxf(mt, xt);
        st = st * __expf(mt - nt) + __expf(xt - nt);
        mt = nt;
    }

    sMs[tid] = ms; sSs[tid] = ss; sMt[tid] = mt; sSt[tid] = st;
    __syncthreads();
    for (int off = 128; off > 0; off >>= 1) {
        if (tid < off) {
            float m1 = sMs[tid], s1 = sSs[tid];
            float m2 = sMs[tid + off], s2 = sSs[tid + off];
            float m  = fmaxf(m1, m2);
            sSs[tid] = s1 * __expf(m1 - m) + s2 * __expf(m2 - m);
            sMs[tid] = m;
            m1 = sMt[tid]; s1 = sSt[tid];
            m2 = sMt[tid + off]; s2 = sSt[tid + off];
            m  = fmaxf(m1, m2);
            sSt[tid] = s1 * __expf(m1 - m) + s2 * __expf(m2 - m);
            sMt[tid] = m;
        }
        __syncthreads();
    }
    __shared__ float shLs, shLt;
    if (tid == 0) {
        shLs = sMs[0] + __logf(sSs[0]);   // log Z_student
        shLt = sMt[0] + __logf(sSt[0]);   // log Z_teacher
    }
    __syncthreads();
    const float Ls = shLs, Lt = shLt;

    float klp = 0.f, klq = 0.f;
    for (int it = 0; it < fullIters; ++it) {
        const int j = (it << 10) + (tid << 2);
        bf16x4 vs = *(const bf16x4*)(ls + j);
        bf16x4 vt = *(const bf16x4*)(lt + j);
        #pragma unroll
        for (int e = 0; e < 4; ++e) {
            float lq = (float)vs[e] - Ls;     // student log-prob
            float lp = (float)vt[e] - Lt;     // teacher log-prob
            float q  = __expf(lq);
            float p  = __expf(lp);
            float m  = 0.5f * q + 0.5f * p;   // beta = 0.5
            float lm = __logf(fmaxf(m, 1e-30f));
            klp += p * (lp - lm);
            klq += q * (lq - lm);
        }
    }
    for (int j = (fullIters << 10) + tid; j < V; j += 256) {
        float lq = (float)ls[j] - Ls;
        float lp = (float)lt[j] - Lt;
        float q  = __expf(lq);
        float p  = __expf(lp);
        float m  = 0.5f * q + 0.5f * p;
        float lm = __logf(fmaxf(m, 1e-30f));
        klp += p * (lp - lm);
        klq += q * (lq - lm);
    }

    sSs[tid] = klp; sSt[tid] = klq;
    __syncthreads();
    for (int off = 128; off > 0; off >>= 1) {
        if (tid < off) { sSs[tid] += sSs[tid + off]; sSt[tid] += sSt[tid + off]; }
        __syncthreads();
    }
    if (tid == 0) {
        float loss = 0.5f * sSs[0] + 0.5f * sSt[0];
        rowLoss[n] = (label[n] != -100) ? loss : 0.f;
    }
}

__global__ __launch_bounds__(256)
void final_reduce(const float* __restrict__ rowLoss,
                  const int* __restrict__ label,
                  float* __restrict__ out, int N)
{
    __shared__ float ssum[256];
    __shared__ int   scnt[256];
    const int tid = threadIdx.x;
    float s = 0.f; int c = 0;
    for (int i = tid; i < N; i += 256) {
        s += rowLoss[i];
        c += (label[i] != -100) ? 1 : 0;
    }
    ssum[tid] = s; scnt[tid] = c;
    __syncthreads();
    for (int off = 128; off > 0; off >>= 1) {
        if (tid < off) { ssum[tid] += ssum[tid + off]; scnt[tid] += scnt[tid + off]; }
        __syncthreads();
    }
    if (tid == 0)
        out[0] = (scnt[0] > 0) ? (ssum[0] / (float)scnt[0]) : 0.0f;
}

// ---------------------------------------------------------------------------
extern "C" void kernel_launch(void* const* d_in, const int* in_sizes, int n_in,
                              void* d_out, int out_size, void* d_ws, size_t ws_size,
                              hipStream_t stream)
{
    const float* Xs  = (const float*)d_in[0];   // student_input [N,H]
    const float* Xt  = (const float*)d_in[1];   // teacher_input [N,H]
    const float* Ws  = (const float*)d_in[2];   // W_student     [V,H]
    const float* Wt  = (const float*)d_in[3];   // W_teacher     [V,H]
    const int*   lbl = (const int*)d_in[4];     // label         [N]

    const int N = in_sizes[4];
    const int H = in_sizes[0] / N;
    const int V = in_sizes[2] / H;

    __bf16* logS   = (__bf16*)d_ws;                    // N*V bf16
    __bf16* logT   = logS + (size_t)N * V;             // N*V bf16
    float* rowLoss = (float*)(logT + (size_t)N * V);   // N f32

    dim3 grid(N / TILE_M, V / TILE_N, 2);   // x = M-tiles: share W tile in L2
    lmhead_gemm_bf16wmma<<<grid, 256, 0, stream>>>(Xs, Ws, Xt, Wt,
                                                   logS, logT, N, V, H);
    jsd_row_kernel<<<N, 256, 0, stream>>>(logS, logT, lbl, rowLoss, V);
    final_reduce<<<1, 256, 0, stream>>>(rowLoss, lbl, (float*)d_out, N);
}
